// DecoderAttn_2302102471316
// MI455X (gfx1250) — matrix operand order repaired and checked by reading, not compiled, not run on hardware
//
#include <hip/hip_runtime.h>
#include <hip/hip_bf16.h>
#include <math.h>

typedef __attribute__((ext_vector_type(16))) __bf16 v16bf;
typedef __attribute__((ext_vector_type(8)))  float  v8f;

#define BM 64
#define BN 128
#define BK 32

__device__ __forceinline__ float sigm(float x) { return 1.0f / (1.0f + __expf(-x)); }

__device__ __forceinline__ float wave_reduce_sum(float v) {
#pragma unroll
  for (int off = 16; off > 0; off >>= 1) v += __shfl_xor(v, off, 32);
  return v;
}

union PkPair { __bf16 h[2]; unsigned int u; };
union Pk4    { __bf16 h[4]; uint2 u2; };
union Frag   { v16bf v; unsigned int u[8]; };
union U8s    { uint4 q; unsigned short s[8]; };

// ---------------------------------------------------------------------------
// f32 -> bf16 bulk convert (weights converted once; reused 51x from L2)
// ---------------------------------------------------------------------------
__global__ void k_cvt_bf16(const float* __restrict__ src, __bf16* __restrict__ dst, int n) {
  const int i = (blockIdx.x * 256 + threadIdx.x) * 8;
  if (i >= n) return;
  const float4 f0 = *reinterpret_cast<const float4*>(src + i);
  const float4 f1 = *reinterpret_cast<const float4*>(src + i + 4);
  Pk4 p0, p1;
  p0.h[0] = (__bf16)f0.x; p0.h[1] = (__bf16)f0.y;
  p0.h[2] = (__bf16)f0.z; p0.h[3] = (__bf16)f0.w;
  p1.h[0] = (__bf16)f1.x; p1.h[1] = (__bf16)f1.y;
  p1.h[2] = (__bf16)f1.z; p1.h[3] = (__bf16)f1.w;
  *reinterpret_cast<uint2*>(dst + i)     = p0.u2;
  *reinterpret_cast<uint2*>(dst + i + 4) = p1.u2;
}

// ---------------------------------------------------------------------------
// C = act( A[M,K](f32) @ B[K,N](bf16) + bias [+ C] ) * row_scale
// 64x128 block tile, 8 waves, wave owns 32x32 (4 WMMAs per K-chunk).
// Double-buffered LDS: one barrier per K-chunk, loads overlap WMMA.
// ---------------------------------------------------------------------------
__global__ __launch_bounds__(256) void k_gemm_bf16(
    const float* __restrict__ A, int lda, const int* __restrict__ amap,
    const __bf16* __restrict__ B, int ldb,
    const float* __restrict__ bias,
    float* __restrict__ C, int ldc, int accumulate,
    const float* __restrict__ row_scale,
    int M, int N, int K, int act /*0=none,1=sigmoid*/) {
  __shared__ __bf16 As[2][BM][BK];  // [m][k]
  __shared__ __bf16 Bs[2][BN][BK];  // [n][k] transposed: frag reads are b128

  const int tid  = threadIdx.x;
  const int lane = tid & 31;
  const int w    = tid >> 5;
  const int gm   = blockIdx.y * BM;
  const int gn   = blockIdx.x * BN;
  const int wm   = (w & 1) * 32;   // 2 waves down
  const int wn   = (w >> 1) * 32;  // 4 waves across
  const int half = lane >> 4;
  const int l16  = lane & 15;

  v8f c00 = {0.f,0.f,0.f,0.f,0.f,0.f,0.f,0.f};
  v8f c01 = c00, c10 = c00, c11 = c00;

  // ---- K-invariant staging setup ----
  const int a_r = tid >> 2;        // 0..63
  const int a_k = (tid & 3) * 8;   // 0,8,16,24
  const int gr  = gm + a_r;
  const int pr  = (gr < M) ? (amap ? amap[gr] : gr) : 0;  // clamp: guarded later
  const float* arow = A + (size_t)pr * lda + a_k;
  const int b_k = (tid >> 4) * 2;  // k pair 0..30
  const int b_n = (tid & 15) * 8;  // 0..120
  const __bf16* bcol = ((gn + b_n) < N) ? (B + (size_t)(gn + b_n)) : B;  // clamp

  auto stageA = [&](int buf, int kk) {
    const float4 f0 = *reinterpret_cast<const float4*>(arow + kk);
    const float4 f1 = *reinterpret_cast<const float4*>(arow + kk + 4);
    Pk4 p0, p1;
    p0.h[0] = (__bf16)f0.x; p0.h[1] = (__bf16)f0.y;
    p0.h[2] = (__bf16)f0.z; p0.h[3] = (__bf16)f0.w;
    p1.h[0] = (__bf16)f1.x; p1.h[1] = (__bf16)f1.y;
    p1.h[2] = (__bf16)f1.z; p1.h[3] = (__bf16)f1.w;
    *reinterpret_cast<uint2*>(&As[buf][a_r][a_k])     = p0.u2;
    *reinterpret_cast<uint2*>(&As[buf][a_r][a_k + 4]) = p1.u2;
  };
  auto stageB = [&](int buf, int kk) {
    const __bf16* p0 = bcol + (size_t)(kk + b_k) * ldb;
    const __bf16* p1 = p0 + ldb;
    if (kk + 2 * BK < K)
      __builtin_prefetch(p0 + (size_t)2 * BK * ldb, 0, 1);
    U8s r0, r1;
    r0.q = *reinterpret_cast<const uint4*>(p0);
    r1.q = *reinterpret_cast<const uint4*>(p1);
#pragma unroll
    for (int j = 0; j < 8; ++j) {
      const unsigned int pk = (unsigned int)r0.s[j] | ((unsigned int)r1.s[j] << 16);
      *reinterpret_cast<unsigned int*>(&Bs[buf][b_n + j][b_k]) = pk;
    }
  };
  auto compute = [&](int buf) {
    Frag a0, a1, b0, b1;
#pragma unroll
    for (int q = 0; q < 8; ++q) {
      const int kk = 2 * q + (q >= 4 ? 8 : 0) + half * 8;  // A 16x32 interleave
      a0.u[q] = *reinterpret_cast<const unsigned int*>(&As[buf][wm + l16][kk]);
      a1.u[q] = *reinterpret_cast<const unsigned int*>(&As[buf][wm + 16 + l16][kk]);
    }
#pragma unroll
    for (int q = 0; q < 8; ++q) {
      const int kb = 2 * q + (q >= 4 ? 8 : 0) + half * 8;  // B: same interleave as A
      b0.u[q] = *reinterpret_cast<const unsigned int*>(&Bs[buf][wn + l16][kb]);
      b1.u[q] = *reinterpret_cast<const unsigned int*>(&Bs[buf][wn + 16 + l16][kb]);
    }
    c00 = __builtin_amdgcn_wmma_f32_16x16x32_bf16(false, a0.v, false, b0.v, (short)0, c00, false, false);
    c01 = __builtin_amdgcn_wmma_f32_16x16x32_bf16(false, a0.v, false, b1.v, (short)0, c01, false, false);
    c10 = __builtin_amdgcn_wmma_f32_16x16x32_bf16(false, a1.v, false, b0.v, (short)0, c10, false, false);
    c11 = __builtin_amdgcn_wmma_f32_16x16x32_bf16(false, a1.v, false, b1.v, (short)0, c11, false, false);
  };

  const int nch = K / BK;  // all K here are multiples of 32
  stageA(0, 0);
  stageB(0, 0);
  __syncthreads();
  for (int ch = 0; ch < nch - 1; ++ch) {
    const int nxt = (ch + 1) & 1;
    stageA(nxt, (ch + 1) * BK);  // issue next chunk's loads first
    stageB(nxt, (ch + 1) * BK);
    compute(ch & 1);             // WMMAs hide the load latency
    __syncthreads();             // single barrier per chunk
  }
  compute((nch - 1) & 1);

  // ---- epilogue: VGPR r -> row (r + 8*half), col = lane&15 ----
#pragma unroll
  for (int mi = 0; mi < 2; ++mi) {
#pragma unroll
    for (int r = 0; r < 8; ++r) {
      const int row = gm + wm + mi * 16 + half * 8 + r;
      if (row >= M) continue;
      const float rs = row_scale ? row_scale[row] : 1.0f;
#pragma unroll
      for (int ni = 0; ni < 2; ++ni) {
        const int col = gn + wn + ni * 16 + l16;
        if (col >= N) continue;
        float v = mi == 0 ? (ni == 0 ? c00[r] : c01[r]) : (ni == 0 ? c10[r] : c11[r]);
        if (bias) v += bias[col];
        const size_t idx = (size_t)row * ldc + col;
        if (accumulate) v += C[idx];
        if (act == 1) v = sigm(v);
        C[idx] = v * rs;
      }
    }
  }
}

// ---------------------------------------------------------------------------
__global__ void k_setup_sort(const int* __restrict__ caps, const int* __restrict__ lens,
                             int* __restrict__ order, int* __restrict__ dl,
                             float* __restrict__ out_caps, float* __restrict__ out_dl) {
  __shared__ int s_len[128];
  const int tid = threadIdx.x;  // 128 threads
  const int l = lens[tid];
  s_len[tid] = l;
  __syncthreads();
  int rank = 0;
  for (int j = 0; j < 128; ++j) {
    const int lj = s_len[j];
    if (lj > l || (lj == l && j < tid)) rank++;
  }
  order[rank] = tid;
  dl[rank] = l - 1;
  out_dl[rank] = (float)(l - 1);
  __syncthreads();
  const int src = order[tid];
  for (int j = 0; j < 52; ++j) out_caps[tid * 52 + j] = (float)caps[src * 52 + j];
}

__global__ void k_rowmap(const int* __restrict__ order, int* __restrict__ rowmap) {
  const int r = blockIdx.x * blockDim.x + threadIdx.x;
  if (r < 128 * 196) rowmap[r] = order[r / 196] * 196 + (r % 196);
}

__global__ void k_mean(const float* __restrict__ feat, const int* __restrict__ order,
                       float* __restrict__ eo) {
  const int e = blockIdx.x * 256 + threadIdx.x;  // grid (8,128)
  const int b = blockIdx.y;
  const float* f = feat + (size_t)order[b] * 196 * 2048 + e;
  float s = 0.0f;
  for (int p = 0; p < 196; ++p) s += f[(size_t)p * 2048];
  eo[b * 2048 + e] = s * (1.0f / 196.0f);
}

// one wave per (b,p): score = relu(enc_proj + dec_proj) . w_full + b_full
__global__ __launch_bounds__(256) void k_att_scores(
    const float* __restrict__ enc_proj, const float* __restrict__ dec_proj,
    const float* __restrict__ w_full, const float* __restrict__ b_full,
    float* __restrict__ scores) {
  const int gw = blockIdx.x * 8 + (threadIdx.x >> 5);  // 3136 blocks -> 25088 waves
  const int lane = threadIdx.x & 31;
  const int b = gw / 196, p = gw % 196;
  const float* ep = enc_proj + ((size_t)b * 196 + p) * 512;
  const float* dp = dec_proj + (size_t)b * 512;
  float s = 0.0f;
#pragma unroll 4
  for (int i = 0; i < 16; ++i) {
    const int a = lane + 32 * i;
    float v = ep[a] + dp[a];
    v = v > 0.0f ? v : 0.0f;
    s += v * w_full[a];
  }
  s = wave_reduce_sum(s);
  if (lane == 0) scores[gw] = s + b_full[0];
}

__global__ __launch_bounds__(256) void k_softmax(
    const float* __restrict__ scores, float* __restrict__ alpha,
    float* __restrict__ out_alphas, const int* __restrict__ dl, int t) {
  __shared__ float red[256];
  const int b = blockIdx.x, tid = threadIdx.x;
  const float s = (tid < 196) ? scores[b * 196 + tid] : -3.0e38f;
  red[tid] = s;
  __syncthreads();
  for (int o = 128; o > 0; o >>= 1) {
    if (tid < o) red[tid] = fmaxf(red[tid], red[tid + o]);
    __syncthreads();
  }
  const float mx = red[0];
  __syncthreads();
  const float e = (tid < 196) ? __expf(s - mx) : 0.0f;
  red[tid] = e;
  __syncthreads();
  for (int o = 128; o > 0; o >>= 1) {
    if (tid < o) red[tid] += red[tid + o];
    __syncthreads();
  }
  const float inv = 1.0f / red[0];
  if (tid < 196) {
    const float a = e * inv;
    alpha[b * 196 + tid] = a;
    const float mask = (dl[b] > t) ? 1.0f : 0.0f;
    out_alphas[((size_t)b * 51 + t) * 196 + tid] = a * mask;
  }
}

// attw[b,e] = sum_p alpha[b,p] * features[order[b],p,e]   (HBM-bound)
__global__ void k_attw(const float* __restrict__ feat, const int* __restrict__ order,
                       const float* __restrict__ alpha, float* __restrict__ attw) {
  const int e = blockIdx.x * 256 + threadIdx.x;  // grid (8,128)
  const int b = blockIdx.y;
  const float* f = feat + (size_t)order[b] * 196 * 2048 + e;
  const float* al = alpha + b * 196;
  float s = 0.0f;
  for (int p = 0; p < 196; ++p) s += al[p] * f[(size_t)p * 2048];
  attw[b * 2048 + e] = s;
}

// x = concat(emb[cap[b,t]], gate * attw)   (128 x 2560)
__global__ void k_build_x(const float* __restrict__ emb, const int* __restrict__ caps,
                          const int* __restrict__ order, const float* __restrict__ gate,
                          const float* __restrict__ attw, float* __restrict__ x, int t) {
  const int col = blockIdx.x * 256 + threadIdx.x;  // grid (10,128)
  const int b = blockIdx.y;
  float v;
  if (col < 512) {
    const int tok = caps[order[b] * 52 + t];
    v = emb[(size_t)tok * 512 + col];
  } else {
    const int j = col - 512;
    v = gate[b * 2048 + j] * attw[b * 2048 + j];
  }
  x[(size_t)b * 2560 + col] = v;
}

__global__ void k_lstm(const float* __restrict__ gates, float* __restrict__ h,
                       float* __restrict__ c, float* __restrict__ hnew,
                       float* __restrict__ maskb, const int* __restrict__ dl, int t) {
  const int col = blockIdx.x * 256 + threadIdx.x;  // grid (2,128)
  const int b = blockIdx.y;
  const float* g = gates + (size_t)b * 2048;
  const float iv = sigm(g[col]);
  const float fv = sigm(g[512 + col]);
  const float gv = tanhf(g[1024 + col]);
  const float ov = sigm(g[1536 + col]);
  const int hi = b * 512 + col;
  const float cn = fv * c[hi] + iv * gv;
  const float hn = ov * tanhf(cn);
  hnew[hi] = hn;
  const float mask = (dl[b] > t) ? 1.0f : 0.0f;
  h[hi] = mask * hn + (1.0f - mask) * h[hi];
  c[hi] = mask * cn + (1.0f - mask) * c[hi];
  if (col == 0) maskb[b] = mask;
}

// ---------------------------------------------------------------------------
extern "C" void kernel_launch(void* const* d_in, const int* in_sizes, int n_in,
                              void* d_out, int out_size, void* d_ws, size_t ws_size,
                              hipStream_t stream) {
  (void)in_sizes; (void)n_in; (void)out_size; (void)ws_size;
  const float* features  = (const float*)d_in[0];
  const int*   captions  = (const int*)d_in[1];
  const int*   cap_len   = (const int*)d_in[2];
  const float* emb       = (const float*)d_in[3];
  const float* W_enc_att = (const float*)d_in[4];
  const float* b_enc_att = (const float*)d_in[5];
  const float* W_dec_att = (const float*)d_in[6];
  const float* b_dec_att = (const float*)d_in[7];
  const float* w_full    = (const float*)d_in[8];
  const float* b_full    = (const float*)d_in[9];
  const float* W_h0      = (const float*)d_in[10];
  const float* b_h0      = (const float*)d_in[11];
  const float* W_c0      = (const float*)d_in[12];
  const float* b_c0      = (const float*)d_in[13];
  const float* W_beta    = (const float*)d_in[14];
  const float* b_beta    = (const float*)d_in[15];
  const float* W_ih      = (const float*)d_in[16];
  const float* b_ih      = (const float*)d_in[17];
  const float* W_hh      = (const float*)d_in[18];
  const float* b_hh      = (const float*)d_in[19];
  const float* W_fc      = (const float*)d_in[20];
  const float* b_fc      = (const float*)d_in[21];

  const int B = 128, P = 196, ENC = 2048, D = 512, V = 10000, T = 51;

  // ---- d_out layout: preds | captions | decode_lengths | alphas (all f32) ----
  float* out_preds  = (float*)d_out;                       // B*T*V
  float* out_caps   = out_preds + (size_t)B * T * V;       // B*52
  float* out_dl     = out_caps + (size_t)B * 52;           // B
  float* out_alphas = out_dl + B;                          // B*T*P

  // ---- workspace carve ----
  char* wp = (char*)d_ws;
  int* order  = (int*)wp;            wp += 128 * sizeof(int);
  int* dl     = (int*)wp;            wp += 128 * sizeof(int);
  int* rowmap = (int*)wp;            wp += (size_t)B * P * sizeof(int);
  float* eo       = (float*)wp;      wp += (size_t)B * ENC * sizeof(float);
  float* h        = (float*)wp;      wp += (size_t)B * D * sizeof(float);
  float* c        = (float*)wp;      wp += (size_t)B * D * sizeof(float);
  float* hnew     = (float*)wp;      wp += (size_t)B * D * sizeof(float);
  float* dec_proj = (float*)wp;      wp += (size_t)B * D * sizeof(float);
  float* gatebuf  = (float*)wp;      wp += (size_t)B * ENC * sizeof(float);
  float* gates    = (float*)wp;      wp += (size_t)B * 4 * D * sizeof(float);
  float* attw     = (float*)wp;      wp += (size_t)B * ENC * sizeof(float);
  float* scores   = (float*)wp;      wp += (size_t)B * P * sizeof(float);
  float* alpha    = (float*)wp;      wp += (size_t)B * P * sizeof(float);
  float* x        = (float*)wp;      wp += (size_t)B * (D + ENC) * sizeof(float);
  float* maskb    = (float*)wp;      wp += (size_t)B * sizeof(float);
  float* enc_proj = (float*)wp;      wp += (size_t)B * P * D * sizeof(float);
  // bf16 weight copies (converted once, streamed from L2 all 51 steps)
  __bf16* wb_enc  = (__bf16*)wp;     wp += (size_t)ENC * D * sizeof(__bf16);
  __bf16* wb_h0   = (__bf16*)wp;     wp += (size_t)ENC * D * sizeof(__bf16);
  __bf16* wb_c0   = (__bf16*)wp;     wp += (size_t)ENC * D * sizeof(__bf16);
  __bf16* wb_dec  = (__bf16*)wp;     wp += (size_t)D * D * sizeof(__bf16);
  __bf16* wb_beta = (__bf16*)wp;     wp += (size_t)D * ENC * sizeof(__bf16);
  __bf16* wb_ih   = (__bf16*)wp;     wp += (size_t)(D + ENC) * 4 * D * sizeof(__bf16);
  __bf16* wb_hh   = (__bf16*)wp;     wp += (size_t)D * 4 * D * sizeof(__bf16);
  __bf16* wb_fc   = (__bf16*)wp;     wp += (size_t)D * V * sizeof(__bf16);

  const int GX_D   = (D + BN - 1) / BN;        // 4
  const int GX_ENC = (ENC + BN - 1) / BN;      // 16
  const int GX_V   = (V + BN - 1) / BN;        // 79
  const int GY_B   = (B + BM - 1) / BM;        // 2
  const int GY_BP  = (B * P + BM - 1) / BM;    // 392

  auto cvt = [&](const float* s, __bf16* d, size_t n) {
    k_cvt_bf16<<<(unsigned)((n / 8 + 255) / 256), 256, 0, stream>>>(s, d, (int)n);
  };

  // ---- setup ----
  k_setup_sort<<<1, 128, 0, stream>>>(captions, cap_len, order, dl, out_caps, out_dl);
  k_rowmap<<<(B * P + 255) / 256, 256, 0, stream>>>(order, rowmap);
  k_mean<<<dim3(8, B), 256, 0, stream>>>(features, order, eo);
  cvt(W_enc_att, wb_enc, (size_t)ENC * D);
  cvt(W_h0, wb_h0, (size_t)ENC * D);
  cvt(W_c0, wb_c0, (size_t)ENC * D);
  cvt(W_dec_att, wb_dec, (size_t)D * D);
  cvt(W_beta, wb_beta, (size_t)D * ENC);
  cvt(W_ih, wb_ih, (size_t)(D + ENC) * 4 * D);
  cvt(W_hh, wb_hh, (size_t)D * 4 * D);
  cvt(W_fc, wb_fc, (size_t)D * V);

  // enc_proj = features_sorted @ W_enc_att + b_enc_att      (25088x2048 @ 2048x512)
  k_gemm_bf16<<<dim3(GX_D, GY_BP), 256, 0, stream>>>(
      features, ENC, rowmap, wb_enc, D, b_enc_att, enc_proj, D, 0, nullptr,
      B * P, D, ENC, 0);
  // h0 / c0
  k_gemm_bf16<<<dim3(GX_D, GY_B), 256, 0, stream>>>(
      eo, ENC, nullptr, wb_h0, D, b_h0, h, D, 0, nullptr, B, D, ENC, 0);
  k_gemm_bf16<<<dim3(GX_D, GY_B), 256, 0, stream>>>(
      eo, ENC, nullptr, wb_c0, D, b_c0, c, D, 0, nullptr, B, D, ENC, 0);

  // ---- sequential decode loop ----
  for (int t = 0; t < T; ++t) {
    // dec_proj = h @ W_dec_att + b
    k_gemm_bf16<<<dim3(GX_D, GY_B), 256, 0, stream>>>(
        h, D, nullptr, wb_dec, D, b_dec_att, dec_proj, D, 0, nullptr, B, D, D, 0);
    k_att_scores<<<(B * P) / 8, 256, 0, stream>>>(enc_proj, dec_proj, w_full, b_full, scores);
    k_softmax<<<B, 256, 0, stream>>>(scores, alpha, out_alphas, dl, t);
    k_attw<<<dim3(8, B), 256, 0, stream>>>(features, order, alpha, attw);
    // gate = sigmoid(h @ W_beta + b_beta)
    k_gemm_bf16<<<dim3(GX_ENC, GY_B), 256, 0, stream>>>(
        h, D, nullptr, wb_beta, ENC, b_beta, gatebuf, ENC, 0, nullptr, B, ENC, D, 1);
    k_build_x<<<dim3(10, B), 256, 0, stream>>>(emb, captions, order, gatebuf, attw, x, t);
    // gates = x @ W_ih + b_ih ; gates += h @ W_hh + b_hh
    k_gemm_bf16<<<dim3(GX_ENC, GY_B), 256, 0, stream>>>(
        x, D + ENC, nullptr, wb_ih, 4 * D, b_ih, gates, 4 * D, 0, nullptr, B, 4 * D, D + ENC, 0);
    k_gemm_bf16<<<dim3(GX_ENC, GY_B), 256, 0, stream>>>(
        h, D, nullptr, wb_hh, 4 * D, b_hh, gates, 4 * D, 1, nullptr, B, 4 * D, D, 0);
    k_lstm<<<dim3(2, B), 256, 0, stream>>>(gates, h, c, hnew, maskb, dl, t);
    // preds = mask * (h_new @ W_fc + b_fc) written strided into d_out
    k_gemm_bf16<<<dim3(GX_V, GY_B), 256, 0, stream>>>(
        hnew, D, nullptr, wb_fc, V, b_fc, out_preds + (size_t)t * V, T * V, 0, maskb,
        B, V, D, 0);
  }
}
